// LorentzMultiheadAttention_80805514707365
// MI455X (gfx1250) — matrix operand order repaired and checked by reading, not compiled, not run on hardware
//
#include <hip/hip_runtime.h>
#include <math.h>

#define D_MODEL 1024
#define NHEADS  16
#define DHEAD   64
#define BATCH   2
#define SEQ     2048
#define BN      (BATCH*SEQ)        /* 4096 rows */
#define KT_PROJ (D_MODEL/32)       /* 32 k-tiles in projection GEMMs */

typedef __attribute__((ext_vector_type(16))) __bf16 v16bf;
typedef __attribute__((ext_vector_type(8)))  float  v8f;
typedef unsigned short u16;
typedef unsigned int   u32;
typedef unsigned long long u64;

// ---------------------------------------------------------------- helpers
__device__ __forceinline__ u16 f2bf(float f) {
    u32 u = __float_as_uint(f);
    u32 r = u + 0x7FFFu + ((u >> 16) & 1u);   // round-to-nearest-even
    return (u16)(r >> 16);
}

// 16-bit A-matrix 16x32 layout (ISA 7.12.2): element e of lane's v16bf -> K index
__device__ __forceinline__ int a_k(int lane, int e) {
    return e + (e < 8 ? 0 : 8) + (lane >= 16 ? 8 : 0);
}

__device__ __forceinline__ v8f wmma_bf16(v16bf a, v16bf b, v8f c) {
    return __builtin_amdgcn_wmma_f32_16x16x32_bf16(false, a, false, b, (short)0, c,
                                                   false, false);
}

// 32B/lane async copy global->LDS (ASYNCcnt path). IOFFSET applies to both the
// LDS and global address (ISA 10.7), so one pair covers a 32B fragment slice.
__device__ __forceinline__ void async_cp32(u32 laddr, const void* g) {
    asm volatile("global_load_async_to_lds_b128 %0, %1, off\n\t"
                 "global_load_async_to_lds_b128 %0, %1, off offset:16"
                 :: "v"(laddr), "v"((u64)(uintptr_t)g) : "memory");
}

// ------------------------------------------------- pack A fragments (generic)
__global__ void pack_a_kernel(const float* __restrict__ src, int ld, int coff,
                              u16* __restrict__ dst, int Mdim, int Kdim) {
    int gid = blockIdx.x * blockDim.x + threadIdx.x;
    if (gid >= Mdim * Kdim) return;
    int tile = gid >> 9, within = gid & 511;
    int lane = within >> 4, e = within & 15;
    int KT = Kdim >> 5;
    int mt = tile / KT, kt = tile % KT;
    int m = lane & 15;
    int k = a_k(lane, e);
    float v = src[(size_t)(mt * 16 + m) * ld + coff + kt * 32 + k];
    dst[gid] = f2bf(v);
}

// --------------------------------------------- pack W^T as B fragments
__global__ void pack_w_kernel(const float* __restrict__ W, u16* __restrict__ dst) {
    int gid = blockIdx.x * blockDim.x + threadIdx.x;
    if (gid >= D_MODEL * D_MODEL) return;
    int tile = gid >> 9, within = gid & 511;
    int lane = within >> 4, e = within & 15;
    int nt = tile / KT_PROJ, kt = tile % KT_PROJ;
    int i = kt * 32 + a_k(lane, e);  // k = input dim
    int o = nt * 16 + (lane & 15);   // n = output dim
    dst[gid] = f2bf(W[(size_t)o * D_MODEL + i]);
}

// ------------------------------------------- projection GEMM  C = A * B + bias
// A: (BN x 1024) bf16 frags; B: (1024 x 1024) bf16 frags; out: f32 row-major
// one wave = 16 rows x 64 cols (4 accs, 5 frag loads / 4 WMMAs per k-step);
// no manual pipelining -- latency hidden by occupancy, avoids register spills.
// 512 blocks x 8 waves = 4096 waves (exact).
__global__ void gemm_proj_kernel(const u16* __restrict__ Afr, const u16* __restrict__ Bfr,
                                 const float* __restrict__ bias, float* __restrict__ out) {
    int wave = (blockIdx.x * blockDim.x + threadIdx.x) >> 5;
    int lane = threadIdx.x & 31;
    int mt  = wave >> 4;             // 0..255
    int nt0 = (wave & 15) * 4;       // 4 n-tiles per wave
    const v16bf* A  = (const v16bf*)Afr;
    const v16bf* Bv = (const v16bf*)Bfr;
    v8f acc0 = {}, acc1 = {}, acc2 = {}, acc3 = {};
    for (int kt = 0; kt < KT_PROJ; ++kt) {
        v16bf a  = A [(size_t)(mt * KT_PROJ + kt) * 32 + lane];
        if (kt + 1 < KT_PROJ)
            __builtin_prefetch(&A[(size_t)(mt * KT_PROJ + kt + 1) * 32 + lane], 0, 1);
        v16bf b0 = Bv[(size_t)((nt0 + 0) * KT_PROJ + kt) * 32 + lane];
        v16bf b1 = Bv[(size_t)((nt0 + 1) * KT_PROJ + kt) * 32 + lane];
        v16bf b2 = Bv[(size_t)((nt0 + 2) * KT_PROJ + kt) * 32 + lane];
        v16bf b3 = Bv[(size_t)((nt0 + 3) * KT_PROJ + kt) * 32 + lane];
        acc0 = wmma_bf16(a, b0, acc0);
        acc1 = wmma_bf16(a, b1, acc1);
        acc2 = wmma_bf16(a, b2, acc2);
        acc3 = wmma_bf16(a, b3, acc3);
    }
    // C layout: VGPR r, lanes 0-15: M=r, N=lane; lanes 16-31: M=r+8, N=lane-16
    int hi = (lane >= 16) ? 8 : 0;
    int n  = lane & 15;
    v8f accs[4] = {acc0, acc1, acc2, acc3};
#pragma unroll
    for (int j = 0; j < 4; ++j) {
        int col = (nt0 + j) * 16 + n;
        float bc = bias[col];
#pragma unroll
        for (int r = 0; r < 8; ++r)
            out[(size_t)(mt * 16 + r + hi) * D_MODEL + col] = accs[j][r] + bc;
    }
}

// --------------------------- pack Q' = [t, s, pad] as A frags per (b,h)
__global__ void pack_q_kernel(const float* __restrict__ Qs, u16* __restrict__ Qp) {
    int gid = blockIdx.x * blockDim.x + threadIdx.x;
    if (gid >= 32 * 128 * 3 * 512) return;
    int tile = gid >> 9, within = gid & 511;
    int lane = within >> 4, e = within & 15;
    int kt = tile % 3; int mt = (tile / 3) & 127; int bh = tile / (3 * 128);
    int b = bh >> 4, h = bh & 15;
    int m = lane & 15;
    int f = kt * 32 + a_k(lane, e);
    const float* srow = Qs + (size_t)(b * SEQ + mt * 16 + m) * D_MODEL + h * DHEAD;
    float v;
    if (f == 0) {
        float s = 1.0f;
        for (int d = 0; d < DHEAD; ++d) { float x = srow[d]; s += x * x; }
        v = sqrtf(s);
    } else if (f <= DHEAD) {
        v = srow[f - 1];
    } else v = 0.0f;
    Qp[gid] = f2bf(v);
}

// --------------------------- pack K'' = [t, -s, pad] as B frags per (b,h)
__global__ void pack_k_kernel(const float* __restrict__ Ks, u16* __restrict__ Kp) {
    int gid = blockIdx.x * blockDim.x + threadIdx.x;
    if (gid >= 32 * 128 * 3 * 512) return;
    int tile = gid >> 9, within = gid & 511;
    int lane = within >> 4, e = within & 15;
    int kt = tile % 3; int jt = (tile / 3) & 127; int bh = tile / (3 * 128);
    int b = bh >> 4, h = bh & 15;
    int f = kt * 32 + a_k(lane, e);
    int j = jt * 16 + (lane & 15);
    const float* srow = Ks + (size_t)(b * SEQ + j) * D_MODEL + h * DHEAD;
    float v;
    if (f == 0) {
        float s = 1.0f;
        for (int d = 0; d < DHEAD; ++d) { float x = srow[d]; s += x * x; }
        v = sqrtf(s);
    } else if (f <= DHEAD) {
        v = -srow[f - 1];
    } else v = 0.0f;
    Kp[gid] = f2bf(v);
}

// --------------------------- pack V spatial as B frags per (b,h)
__global__ void pack_v_kernel(const float* __restrict__ Vs, u16* __restrict__ Vp) {
    int gid = blockIdx.x * blockDim.x + threadIdx.x;
    if (gid >= 32 * 64 * 4 * 512) return;
    int tile = gid >> 9, within = gid & 511;
    int lane = within >> 4, e = within & 15;
    int nt = tile & 3; int jt = (tile >> 2) & 63; int bh = tile / (4 * 64);
    int b = bh >> 4, h = bh & 15;
    int j = jt * 32 + a_k(lane, e);
    int d = nt * 16 + (lane & 15);
    Vp[gid] = f2bf(Vs[(size_t)(b * SEQ + j) * D_MODEL + h * DHEAD + d]);
}

// ----------------------------------------------- flash attention
// one wave = 16 query rows; K/V fragments double-buffered in LDS via
// global_load_async_to_lds_b128 (ASYNCcnt), 10 fragments (6 K + 4 V) per step.
#define AW     4                      /* waves per block */
#define NFRAG  10                     /* fragments per j-step */
#define FRAGU  512                    /* u16 per fragment */
__global__ void attn_kernel(const u16* __restrict__ Qp, const u16* __restrict__ Kp,
                            const u16* __restrict__ Vp, float* __restrict__ Z) {
    __shared__ __align__(32) u16 kvbuf[AW][2][NFRAG * FRAGU];  // 80 KB
    __shared__ __align__(32) u16 pst[AW][32][16];              // P A-frag stage
    int lane = threadIdx.x & 31;
    int w    = threadIdx.x >> 5;
    int wave = blockIdx.x * AW + w;
    int mt = wave & 127;
    int bh = wave >> 7;
    int b = bh >> 4, h = bh & 15;
    const v16bf* Qv = (const v16bf*)Qp;

    size_t qb = ((size_t)bh * 128 + mt) * 3;
    v16bf qa0 = Qv[(qb + 0) * 32 + lane];
    v16bf qa1 = Qv[(qb + 1) * 32 + lane];
    v16bf qa2 = Qv[(qb + 2) * 32 + lane];

    v8f acc[4];
#pragma unroll
    for (int d = 0; d < 4; ++d) acc[d] = (v8f){};
    float mst[8], lst[8];
    const float NEG_INF = -__builtin_inff();
#pragma unroll
    for (int r = 0; r < 8; ++r) { mst[r] = NEG_INF; lst[r] = 0.0f; }

    const float SCL = 0.125f;        // 1/sqrt(64)
    int m_hi = (lane >= 16) ? 8 : 0;
    int n    = lane & 15;

    // issue async fragment loads for step jt into LDS buffer bsel
    auto issue = [&](int jt, int bsel) {
        u16* dstb = &kvbuf[w][bsel][0];
        size_t kb = ((size_t)bh * 128 + jt * 2) * 3;
        size_t vb = ((size_t)bh * 64 + jt) * 4;
#pragma unroll
        for (int f = 0; f < 6; ++f)
            async_cp32((u32)(uintptr_t)(dstb + f * FRAGU + lane * 16),
                       Kp + ((kb + f) * 32 + lane) * 16);
#pragma unroll
        for (int f = 0; f < 4; ++f)
            async_cp32((u32)(uintptr_t)(dstb + (6 + f) * FRAGU + lane * 16),
                       Vp + ((vb + f) * 32 + lane) * 16);
    };

    issue(0, 0);
    for (int jt32 = 0; jt32 < SEQ / 32; ++jt32) {
        int cur = jt32 & 1;
        if (jt32 + 1 < SEQ / 32) {
            issue(jt32 + 1, cur ^ 1);
            // async loads retire in order: <=20 outstanding => current buffer done
            asm volatile("s_wait_asynccnt %0" :: "i"(2 * NFRAG) : "memory");
        } else {
            asm volatile("s_wait_asynccnt 0" ::: "memory");
        }
        const u16* bufb = &kvbuf[w][cur][0];

        // S for 32 keys = two 16x16 tiles (K-dim 96 -> 3 WMMAs each)
        v8f s0 = {}, s1 = {};
        v16bf k0 = *(const v16bf*)(bufb + 0 * FRAGU + lane * 16);
        v16bf k1 = *(const v16bf*)(bufb + 1 * FRAGU + lane * 16);
        v16bf k2 = *(const v16bf*)(bufb + 2 * FRAGU + lane * 16);
        s0 = wmma_bf16(qa0, k0, s0);
        s0 = wmma_bf16(qa1, k1, s0);
        s0 = wmma_bf16(qa2, k2, s0);
        v16bf k3 = *(const v16bf*)(bufb + 3 * FRAGU + lane * 16);
        v16bf k4 = *(const v16bf*)(bufb + 4 * FRAGU + lane * 16);
        v16bf k5 = *(const v16bf*)(bufb + 5 * FRAGU + lane * 16);
        s1 = wmma_bf16(qa0, k3, s1);
        s1 = wmma_bf16(qa1, k4, s1);
        s1 = wmma_bf16(qa2, k5, s1);

        float alpha[8];
#pragma unroll
        for (int r = 0; r < 8; ++r) {
            float v0 = s0[r] * SCL;
            float v1 = s1[r] * SCL;
            float vm = fmaxf(v0, v1);
            vm = fmaxf(vm, __shfl_xor(vm, 8, 32));
            vm = fmaxf(vm, __shfl_xor(vm, 4, 32));
            vm = fmaxf(vm, __shfl_xor(vm, 2, 32));
            vm = fmaxf(vm, __shfl_xor(vm, 1, 32));
            float mold = mst[r];
            float mnew = fmaxf(mold, vm);
            float a = (mold == NEG_INF) ? 0.0f : __expf(mold - mnew);
            mst[r] = mnew; alpha[r] = a;
            float p0 = __expf(v0 - mnew);
            float p1 = __expf(v1 - mnew);
            float rs = p0 + p1;
            rs += __shfl_xor(rs, 8, 32);
            rs += __shfl_xor(rs, 4, 32);
            rs += __shfl_xor(rs, 2, 32);
            rs += __shfl_xor(rs, 1, 32);
            lst[r] = lst[r] * a + rs;
            // scatter P into A-fragment layout in LDS (row m, kcol = tile*16 + n)
            int mrow = r + m_hi;
            {   int k = n;
                pst[w][mrow + 16 * ((k >> 3) & 1)][(k & 7) | ((k & 16) >> 1)] = f2bf(p0); }
            {   int k = 16 + n;
                pst[w][mrow + 16 * ((k >> 3) & 1)][(k & 7) | ((k & 16) >> 1)] = f2bf(p1); }
        }
#pragma unroll
        for (int d = 0; d < 4; ++d)
#pragma unroll
            for (int r = 0; r < 8; ++r) acc[d][r] *= alpha[r];

        asm volatile("s_wait_dscnt 0" ::: "memory");   // wave-local LDS RAW fence
        v16bf pa = *(const v16bf*)&pst[w][lane][0];

#pragma unroll
        for (int d = 0; d < 4; ++d) {
            v16bf bv = *(const v16bf*)(bufb + (6 + d) * FRAGU + lane * 16);
            acc[d] = wmma_bf16(pa, bv, acc[d]);
        }
    }
    // normalize and write merged-head layout
#pragma unroll
    for (int r = 0; r < 8; ++r) {
        float inv = 1.0f / lst[r];
        int row = b * SEQ + mt * 16 + r + m_hi;
#pragma unroll
        for (int d = 0; d < 4; ++d)
            Z[(size_t)row * D_MODEL + h * DHEAD + d * 16 + n] = acc[d][r] * inv;
    }
}

// ---------------------------------------------- finalize: time-concat output
__global__ void finalize_kernel(const float* __restrict__ O, float* __restrict__ out) {
    int lane = threadIdx.x & 31;
    int row  = blockIdx.x * 8 + (threadIdx.x >> 5);
    const float* src = O + (size_t)row * D_MODEL;
    float s = 0.0f;
    for (int c = lane; c < D_MODEL; c += 32) { float v = src[c]; s += v * v; }
    s += __shfl_xor(s, 16, 32);
    s += __shfl_xor(s,  8, 32);
    s += __shfl_xor(s,  4, 32);
    s += __shfl_xor(s,  2, 32);
    s += __shfl_xor(s,  1, 32);
    float* orow = out + (size_t)row * (D_MODEL + 1);
    if (lane == 0) orow[0] = sqrtf(1.0f + s);
    for (int c = lane; c < D_MODEL; c += 32) orow[1 + c] = src[c];
}

// ------------------------------------------------------------------ launcher
extern "C" void kernel_launch(void* const* d_in, const int* in_sizes, int n_in,
                              void* d_out, int out_size, void* d_ws, size_t ws_size,
                              hipStream_t stream) {
    const float* x  = (const float*)d_in[0];
    const float* Wq = (const float*)d_in[1];
    const float* bq = (const float*)d_in[2];
    const float* Wk = (const float*)d_in[3];
    const float* bk = (const float*)d_in[4];
    const float* Wv = (const float*)d_in[5];
    const float* bv = (const float*)d_in[6];
    const float* Wo = (const float*)d_in[7];
    const float* bo = (const float*)d_in[8];
    float* out = (float*)d_out;
    char* ws = (char*)d_ws;

    const size_t MB = 1u << 20;
    u16*   XA = (u16*)(ws + 0);            // 8 MB  bf16 A-frags of x spatial
    u16*   WB[4];
    for (int i = 0; i < 4; ++i) WB[i] = (u16*)(ws + 8 * MB + i * 2 * MB);  // 8 MB
    float* Qs = (float*)(ws + 16 * MB);    // 16 MB f32 (BNx1024)
    float* Ks = (float*)(ws + 32 * MB);    // 16 MB
    float* Vs = (float*)(ws + 48 * MB);    // 16 MB
    u16*   Qp = (u16*)(ws + 64 * MB);      // 12 MB
    u16*   Kp = (u16*)(ws + 76 * MB);      // 12 MB
    u16*   Vp = (u16*)(ws + 88 * MB);      // 8 MB  (peak 96 MB)
    float* Z  = (float*)(ws + 16 * MB);    // reuse Qs region
    u16*   ZA = (u16*)(ws + 32 * MB);      // reuse Ks region
    float* Of = (float*)(ws + 40 * MB);    // reuse Ks tail + Vs

    dim3 blk(256);
    // pack inputs
    pack_a_kernel<<<(BN * D_MODEL + 255) / 256, blk, 0, stream>>>(x, D_MODEL + 1, 1,
                                                                  XA, BN, D_MODEL);
    pack_w_kernel<<<(D_MODEL * D_MODEL + 255) / 256, blk, 0, stream>>>(Wq, WB[0]);
    pack_w_kernel<<<(D_MODEL * D_MODEL + 255) / 256, blk, 0, stream>>>(Wk, WB[1]);
    pack_w_kernel<<<(D_MODEL * D_MODEL + 255) / 256, blk, 0, stream>>>(Wv, WB[2]);
    pack_w_kernel<<<(D_MODEL * D_MODEL + 255) / 256, blk, 0, stream>>>(Wo, WB[3]);
    // Q/K/V projections (WMMA, 16x64 per wave, occupancy-pipelined)
    gemm_proj_kernel<<<512, blk, 0, stream>>>(XA, WB[0], bq, Qs);
    gemm_proj_kernel<<<512, blk, 0, stream>>>(XA, WB[1], bk, Ks);
    gemm_proj_kernel<<<512, blk, 0, stream>>>(XA, WB[2], bv, Vs);
    // pack per-head Lorentz fragments
    pack_q_kernel<<<(32 * 128 * 3 * 512 + 255) / 256, blk, 0, stream>>>(Qs, Qp);
    pack_k_kernel<<<(32 * 128 * 3 * 512 + 255) / 256, blk, 0, stream>>>(Ks, Kp);
    pack_v_kernel<<<(32 * 64 * 4 * 512 + 255) / 256, blk, 0, stream>>>(Vs, Vp);
    // flash attention (WMMA + async double-buffered K/V)
    attn_kernel<<<(32 * 128) / AW, AW * 32, 0, stream>>>(Qp, Kp, Vp, Z);
    // output projection (WMMA)
    pack_a_kernel<<<(BN * D_MODEL + 255) / 256, blk, 0, stream>>>(Z, D_MODEL, 0,
                                                                  ZA, BN, D_MODEL);
    gemm_proj_kernel<<<512, blk, 0, stream>>>(ZA, WB[3], bo, Of);
    // time-concat final output (BN x 1025)
    finalize_kernel<<<BN / 8, blk, 0, stream>>>(Of, out);
}
